// Gate_80410377716149
// MI455X (gfx1250) — compile-verified, matrix operand order checked by reading, not run on hardware
//
#include <hip/hip_runtime.h>
#include <hip/hip_bf16.h>
#include <math.h>

// ---- problem constants (from reference) ----
#define T_TOKENS   16384
#define DIM        7168
#define N_EXPERTS  256
#define TOPK       8
#define N_GROUPS   8
#define GROUP_SIZE 32
#define TOPK_GROUPS 4
#define ROUTE_SCALE 2.5f

// ---- CDNA5 WMMA / TDM vector types ----
typedef __attribute__((ext_vector_type(16))) __bf16        v16bf;
typedef __attribute__((ext_vector_type(8)))  float         v8f;
typedef __attribute__((ext_vector_type(4)))  unsigned int  v4u_;
typedef __attribute__((ext_vector_type(8)))  int           v8i_;
typedef __attribute__((ext_vector_type(4)))  int           v4i_;

// 16/32-byte PODs (trivially copyable for __builtin_bit_cast)
struct alignas(16) U4    { unsigned int x, y, z, w; };
struct alignas(16) F4    { float x, y, z, w; };
struct alignas(32) U32x8 { U4 lo, hi; };          // 16 bf16 = one WMMA fragment
struct alignas(16) BF8   { __bf16 v[8]; };

#if __has_builtin(__builtin_amdgcn_tensor_load_to_lds)
#define HAS_TDM 1
#else
#define HAS_TDM 0
#endif

// ============================================================================
// Kernel 0: one-shot weight conversion fp32 -> bf16 (hardware cvt, b128 stores)
// ============================================================================
__global__ __launch_bounds__(256) void convert_w_bf16(
    const float* __restrict__ w, BF8* __restrict__ out) {
  const int i = blockIdx.x * blockDim.x + threadIdx.x;  // 8 elements per thread
  const F4* src = reinterpret_cast<const F4*>(w) + (size_t)i * 2;
  F4 f0 = src[0], f1 = src[1];
  BF8 o;
  o.v[0] = (__bf16)f0.x; o.v[1] = (__bf16)f0.y;
  o.v[2] = (__bf16)f0.z; o.v[3] = (__bf16)f0.w;
  o.v[4] = (__bf16)f1.x; o.v[5] = (__bf16)f1.y;
  o.v[6] = (__bf16)f1.z; o.v[7] = (__bf16)f1.w;
  out[i] = o;
}

// ----------------------------------------------------------------------------
// TDM: DMA one [256 expert x 32 k] bf16 slab of the weight tensor into LDS.
// D# per CDNA5 ISA:
//   group0: count=1 | lds_addr | global_addr | type=2
//   group1: data_size=2B, tensor_dim0=7168, tensor_dim1=256,
//           tile_dim0=32 (k), tile_dim1=256 (experts), dim0_stride=7168
// LDS result is row-major [expert][k0..31] bf16 == fragment-ready layout.
// This toolchain exposes the 6-arg builtin:
//   (uint32x4 g0, int32x8 g1, int32x4 g2, int32x4 g3, int32x8 gx, i32 cpol)
// ----------------------------------------------------------------------------
#if HAS_TDM
static __device__ __forceinline__ void tdm_load_weight_tile(
    const __bf16* wbf, unsigned lds_off, int kb) {
  unsigned long long ga =
      (unsigned long long)(uintptr_t)wbf + (unsigned long long)kb * 2ull;
  v4u_ g0;
  g0.x = 1u;                                                  // count=1 (valid)
  g0.y = lds_off;                                             // lds_addr
  g0.z = (unsigned)ga;                                        // global_addr lo
  g0.w = (unsigned)((ga >> 32) & 0x01FFFFFFull) | (2u << 30); // addr hi | type=2
  v8i_ g1;
  g1[0] = 0x00010000;                        // data_size=1 (2 bytes), mask=0
  g1[1] = (int)((DIM & 0xFFFF) << 16);       // tensor_dim0[15:0]  @ bits 63:48
  g1[2] = (int)((N_EXPERTS & 0xFFFF) << 16); // tensor_dim1[15:0]  @ bits 95:80
  g1[3] = (int)(32u << 16);                  // tile_dim0 = 32     @ bits 127:112
  g1[4] = N_EXPERTS;                         // tile_dim1 = 256    @ bits 143:128
  g1[5] = DIM;                               // tensor_dim0_stride[31:0]
  g1[6] = 0;
  g1[7] = 0;
  v4i_ z4 = {0, 0, 0, 0};
  v8i_ z8 = {0, 0, 0, 0, 0, 0, 0, 0};
  __builtin_amdgcn_tensor_load_to_lds(g0, g1, z4, z4, z8, 0);
}
#endif

// A fragment: 16-bit A layout -> lane-half kh holds K {8kh..8kh+7, 16+8kh..23+8kh}
static __device__ __forceinline__ v16bf load_a_frag(
    const float* __restrict__ xrow, int kb, int kh) {
  const F4* x4 = reinterpret_cast<const F4*>(xrow + kb + 8 * kh);
  F4 a0 = x4[0], a1 = x4[1];   // K = 8kh .. 8kh+7
  F4 a2 = x4[4], a3 = x4[5];   // K = 16+8kh .. 23+8kh
  v16bf a;
  a[0]  = (__bf16)a0.x; a[1]  = (__bf16)a0.y; a[2]  = (__bf16)a0.z; a[3]  = (__bf16)a0.w;
  a[4]  = (__bf16)a1.x; a[5]  = (__bf16)a1.y; a[6]  = (__bf16)a1.z; a[7]  = (__bf16)a1.w;
  a[8]  = (__bf16)a2.x; a[9]  = (__bf16)a2.y; a[10] = (__bf16)a2.z; a[11] = (__bf16)a2.w;
  a[12] = (__bf16)a3.x; a[13] = (__bf16)a3.y; a[14] = (__bf16)a3.z; a[15] = (__bf16)a3.w;
  return a;
}

// ============================================================================
// Kernel 1: scores = sigmoid(x @ W^T) via v_wmma_f32_16x16x32_bf16
//   grid = 128 blocks x 256 threads (8 wave32)
//   block tile: 128 tokens x 256 experts; wave tile: 32M x 128N (16 acc)
//   weight slab TDM-DMA'd into double-buffered LDS; each B fragment feeds 2 WMMAs
// ============================================================================
__global__ __launch_bounds__(256) void gate_scores_wmma(
    const float* __restrict__ x, const __bf16* __restrict__ wbf,
    float* __restrict__ scores) {
  __shared__ U4 Bs[2][N_EXPERTS * 4];   // 2 x 16 KB, [buf][expert][k-chunk]

  const int tid   = threadIdx.x;
  const int wave  = tid >> 5;
  const int lane  = tid & 31;
  const int ln    = lane & 15;
  const int kh    = lane >> 4;
  const int mwave = wave >> 1;          // 0..3 -> 32-token sub-tile
  const int nhalf = wave & 1;           // 0..1 -> expert half

  const int row0 = blockIdx.x * 128 + mwave * 32 + ln;
  const float* xrow0 = x + (size_t)row0 * DIM;
  const float* xrow1 = xrow0 + (size_t)16 * DIM;

  const unsigned lds_base = (unsigned)(uintptr_t)&Bs[0][0];

#if HAS_TDM
  if (wave == 0) {                       // prologue: DMA tile 0 into buffer 0
    tdm_load_weight_tile(wbf, lds_base, 0);
    __builtin_amdgcn_s_wait_tensorcnt(0);
  }
  __syncthreads();
#endif

  v8f acc[2][8];
  #pragma unroll
  for (int f = 0; f < 2; ++f)
    #pragma unroll
    for (int i = 0; i < 8; ++i)
      acc[f][i] = (v8f){0.f,0.f,0.f,0.f,0.f,0.f,0.f,0.f};

  int buf = 0;
  for (int kb = 0; kb < DIM; kb += 32, buf ^= 1) {
#if HAS_TDM
    if (wave == 0 && kb + 32 < DIM)      // DMA next tile while we compute
      tdm_load_weight_tile(wbf, lds_base + (unsigned)((buf ^ 1) * 16384), kb + 32);
#else
    // fallback: cooperative bf16 copy, single buffer
    buf = 0;
    __syncthreads();
    {
      const U4* src = reinterpret_cast<const U4*>(wbf + (size_t)tid * DIM + kb);
      #pragma unroll
      for (int c = 0; c < 4; ++c) Bs[0][tid * 4 + c] = src[c];
    }
    __syncthreads();
#endif

    v16bf a0 = load_a_frag(xrow0, kb, kh);
    v16bf a1 = load_a_frag(xrow1, kb, kh);
    if (kb + 32 < DIM) __builtin_prefetch(xrow0 + kb + 32, 0, 3);

    #pragma unroll
    for (int nt = 0; nt < 8; ++nt) {
      const int n = nhalf * 128 + nt * 16 + ln;
      U32x8 bu;
      bu.lo = Bs[buf][n * 4 + kh];
      bu.hi = Bs[buf][n * 4 + 2 + kh];
      v16bf bfrag = __builtin_bit_cast(v16bf, bu);
      acc[0][nt] = __builtin_amdgcn_wmma_f32_16x16x32_bf16(
          false, a0, false, bfrag, (short)0, acc[0][nt], false, false);
      acc[1][nt] = __builtin_amdgcn_wmma_f32_16x16x32_bf16(
          false, a1, false, bfrag, (short)0, acc[1][nt], false, false);
    }

#if HAS_TDM
    if (wave == 0 && kb + 32 < DIM)
      __builtin_amdgcn_s_wait_tensorcnt(0);  // next buffer is fully in LDS
    __syncthreads();                          // release all waves into next iter
#endif
  }

  // epilogue: sigmoid + store fp32 scores
  // C layout: lanes 0-15 -> M=r, lanes 16-31 -> M=r+8; N=ln within tile
  #pragma unroll
  for (int f = 0; f < 2; ++f) {
    const int token_base = blockIdx.x * 128 + mwave * 32 + f * 16 + 8 * kh;
    #pragma unroll
    for (int nt = 0; nt < 8; ++nt) {
      const int n = nhalf * 128 + nt * 16 + ln;
      #pragma unroll
      for (int r = 0; r < 8; ++r) {
        float v = acc[f][nt][r];
        float sg = 1.0f / (1.0f + __expf(-v));
        scores[(size_t)(token_base + r) * N_EXPERTS + n] = sg;
      }
    }
  }
}

// ============================================================================
// Kernel 2: grouped top-k routing, one thread per token
// ============================================================================
__global__ __launch_bounds__(256) void gate_topk(
    const float* __restrict__ scores, const float* __restrict__ bias,
    float* __restrict__ w_out, int* __restrict__ i_out) {
  __shared__ float bs[N_EXPERTS];
  bs[threadIdx.x] = bias[threadIdx.x];
  __syncthreads();

  const int t = blockIdx.x * blockDim.x + threadIdx.x;
  const float* s = scores + (size_t)t * N_EXPERTS;

  float gsc[N_GROUPS];
  #pragma unroll
  for (int g = 0; g < N_GROUPS; ++g) {
    float m1 = -INFINITY, m2 = -INFINITY;
    for (int j = 0; j < GROUP_SIZE; ++j) {
      const int e = g * GROUP_SIZE + j;
      const float v = s[e] + bs[e];
      if (v > m1)      { m2 = m1; m1 = v; }
      else if (v > m2) { m2 = v; }
    }
    gsc[g] = m1 + m2;
  }

  unsigned sel = 0;
  for (int r = 0; r < TOPK_GROUPS; ++r) {
    int best = -1; float bv = -INFINITY;
    #pragma unroll
    for (int g = 0; g < N_GROUPS; ++g)
      if (!((sel >> g) & 1u) && gsc[g] > bv) { bv = gsc[g]; best = g; }
    if (best >= 0) sel |= 1u << best;
  }

  float tv[TOPK]; int ti[TOPK];
  #pragma unroll
  for (int k = 0; k < TOPK; ++k) { tv[k] = -INFINITY; ti[k] = 0; }
  for (int g = 0; g < N_GROUPS; ++g) {
    if (!((sel >> g) & 1u)) continue;
    for (int j = 0; j < GROUP_SIZE; ++j) {
      const int e = g * GROUP_SIZE + j;
      const float v = s[e] + bs[e];
      if (v > tv[TOPK - 1]) {
        tv[TOPK - 1] = v; ti[TOPK - 1] = e;
        #pragma unroll
        for (int p = TOPK - 1; p > 0; --p) {
          if (tv[p] > tv[p - 1]) {   // strict: stable, lower index wins ties
            float tf = tv[p]; tv[p] = tv[p - 1]; tv[p - 1] = tf;
            int   tn = ti[p]; ti[p] = ti[p - 1]; ti[p - 1] = tn;
          }
        }
      }
    }
  }

  float wv[TOPK]; float wsum = 0.f;
  #pragma unroll
  for (int k = 0; k < TOPK; ++k) { wv[k] = s[ti[k]]; wsum += wv[k]; }
  const float inv = ROUTE_SCALE / wsum;
  #pragma unroll
  for (int k = 0; k < TOPK; ++k) {
    w_out[(size_t)t * TOPK + k] = wv[k] * inv;
    i_out[(size_t)t * TOPK + k] = ti[k];
  }
}

// ============================================================================
extern "C" void kernel_launch(void* const* d_in, const int* in_sizes, int n_in,
                              void* d_out, int out_size, void* d_ws, size_t ws_size,
                              hipStream_t stream) {
  const float* x    = (const float*)d_in[0];   // [16384, 7168]
  const float* w    = (const float*)d_in[1];   // [256, 7168]
  const float* bias = (const float*)d_in[2];   // [256]

  // workspace layout: [0,16MB) fp32 scores, [16MB, +3.7MB) bf16 weight
  float*  scores = (float*)d_ws;
  __bf16* wbf    = (__bf16*)((char*)d_ws + (size_t)T_TOKENS * N_EXPERTS * 4);

  float* w_out = (float*)d_out;                                    // [16384,8] f32
  int*   i_out = (int*)((float*)d_out + (size_t)T_TOKENS * TOPK);  // [16384,8] i32

  convert_w_bf16<<<(N_EXPERTS * DIM) / (256 * 8), 256, 0, stream>>>(w, (BF8*)wbf);
  gate_scores_wmma<<<T_TOKENS / 128, 256, 0, stream>>>(x, wbf, scores);
  gate_topk<<<T_TOKENS / 256, 256, 0, stream>>>(scores, bias, w_out, i_out);
}
